// H3_56195352101229
// MI455X (gfx1250) — matrix operand order repaired and checked by reading, not compiled, not run on hardware
//
#include <hip/hip_runtime.h>

// H3 layer forward for MI455X (gfx1250, wave32, WMMA bf16 16x16x32).
// cvt->bf16 | QKV wmma-GEMM (2Mx4N blocked) | shift-SSM causal conv (VALU) |
// kv outer product | diag-SSM = 128 triangular-Toeplitz GEMMs (wmma, LDS pair
// table for A, 2Mx4N) | mul_sum combine | output wmma-GEMM (2Mx4N blocked).
// 2Mx4N chosen so acc(64 VGPR) + live frags(48 VGPR) stays well under the 256
// directly-addressable VGPRs -- round 2's 4Mx4N spilled accumulators to scratch.

#define L_SEQ 2048
#define DMODEL 1024
#define NHEAD 8
#define HGRP 128

typedef __attribute__((ext_vector_type(16))) __bf16 v16bf;
typedef __attribute__((ext_vector_type(8)))  float  v8f;

union Frag { v16bf v; unsigned int u[8]; unsigned short h[16]; };

__device__ __forceinline__ unsigned short f2bf(float f) {
  unsigned int x = __builtin_bit_cast(unsigned int, f);
  x += 0x7fffu + ((x >> 16) & 1u);              // round-to-nearest-even
  return (unsigned short)(x >> 16);
}
__device__ __forceinline__ float bf2f(unsigned short h) {
  return __builtin_bit_cast(float, (unsigned int)h << 16);
}
__device__ __forceinline__ v8f wmma_bf16(v16bf a, v16bf b, v8f c) {
  return __builtin_amdgcn_wmma_f32_16x16x32_bf16(false, a, false, b, (short)0, c,
                                                 false, false);
}
// A-matrix slot->K offset (contiguous order, matches B fills):
// lanes 0-15 hold K {0-15}, lanes 16-31 hold K {16-31}, 2 K per VGPR.
__device__ __forceinline__ int akk(int j, int hi) {
  return 2 * j + 16 * hi;
}

// ---------------- fp32 -> bf16 convert ----------------
__global__ void cvt_bf16(const float* __restrict__ src,
                         unsigned short* __restrict__ dst, int n) {
  int i = blockIdx.x * blockDim.x + threadIdx.x;
  int stride = gridDim.x * blockDim.x;
  for (; i < n; i += stride) dst[i] = f2bf(src[i]);
}

// ---------------- QKV: out[e,t] = sum_d u[t,d]*W[e,d] + b[e] ----------------
// M=t, N=e, K=d. Each wave: 32x64 output (2x4 wmma tiles), 1024 waves/matrix.
__global__ void __launch_bounds__(256, 1)
qkv_gemm(const unsigned short* __restrict__ uA,
         const unsigned short* __restrict__ Wq,
         const unsigned short* __restrict__ Wk,
         const unsigned short* __restrict__ Wv,
         const float* __restrict__ bq, const float* __restrict__ bk,
         const float* __restrict__ bv,
         float* __restrict__ qo, float* __restrict__ ko,
         float* __restrict__ vo) {
  const unsigned short* W = (blockIdx.y == 0) ? Wq : (blockIdx.y == 1 ? Wk : Wv);
  const float* bias = (blockIdx.y == 0) ? bq : (blockIdx.y == 1 ? bk : bv);
  float* out = (blockIdx.y == 0) ? qo : (blockIdx.y == 1 ? ko : vo);
  int wave = threadIdx.x >> 5;
  int wid = blockIdx.x * 8 + wave;              // 1024 waves
  int t0 = (wid & 63) * 32;                     // 64 t-groups of 32
  int e0 = (wid >> 6) * 64;                     // 16 e-groups of 64
  int lane = threadIdx.x & 31, mrow = lane & 15, hi = lane >> 4;
  v8f acc[2][4] = {};
  const unsigned short* ar[2];
  const unsigned short* br[4];
#pragma unroll
  for (int i = 0; i < 2; ++i) ar[i] = uA + (t0 + 16 * i + mrow) * DMODEL;
#pragma unroll
  for (int i = 0; i < 4; ++i) br[i] = W + (e0 + 16 * i + mrow) * DMODEL;
  for (int k0 = 0; k0 < DMODEL; k0 += 32) {
    __builtin_prefetch(ar[0] + k0 + 256);       // -> global_prefetch_b8
    __builtin_prefetch(br[0] + k0 + 256);
    Frag a[2], b[4];
#pragma unroll
    for (int i = 0; i < 2; ++i)
#pragma unroll
      for (int j = 0; j < 8; ++j)
        a[i].u[j] = *(const unsigned int*)(ar[i] + k0 + akk(j, hi));
#pragma unroll
    for (int i = 0; i < 4; ++i)
#pragma unroll
      for (int j = 0; j < 8; ++j)
        b[i].u[j] = *(const unsigned int*)(br[i] + k0 + 2 * j + 16 * hi);
#pragma unroll
    for (int mi = 0; mi < 2; ++mi)
#pragma unroll
      for (int ct = 0; ct < 4; ++ct)
        acc[mi][ct] = wmma_bf16(a[mi].v, b[ct].v, acc[mi][ct]);
  }
#pragma unroll
  for (int ct = 0; ct < 4; ++ct) {
    float bb = bias[e0 + 16 * ct + mrow];
#pragma unroll
    for (int mi = 0; mi < 2; ++mi) {
      float* op = out + (e0 + 16 * ct + mrow) * L_SEQ + t0 + 16 * mi + 8 * hi;
#pragma unroll
      for (int r = 0; r < 8; ++r) op[r] = acc[mi][ct][r] + bb;
    }
  }
}

// -------- shift-SSM: k'[c,t] = sum_{s<=t} kern[c,t-s]*k[c,s] + D[c]*k[c,t] ------
__global__ void shift_conv(const float* __restrict__ kin,
                           const float* __restrict__ kernk,
                           const float* __restrict__ kD,
                           float* __restrict__ kout) {
  int c = blockIdx.x;
  int t0 = blockIdx.y * 256;
  int tid = threadIdx.x;
  int t = t0 + tid;
  __shared__ float ks[256];
  __shared__ float kw[512];
  const float* kc = kin + c * L_SEQ;
  const float* kk = kernk + c * L_SEQ;
  float acc = 0.f;
  for (int s0 = 0; s0 <= t0; s0 += 256) {
    int base = t0 - s0 - 255;
    ks[tid] = kc[s0 + tid];
    int i0 = base + tid, i1 = base + tid + 256;
    kw[tid] = (i0 >= 0 && i0 < L_SEQ) ? kk[i0] : 0.f;
    kw[tid + 256] = (i1 >= 0 && i1 < L_SEQ) ? kk[i1] : 0.f;
    __syncthreads();
#pragma unroll 8
    for (int s = 0; s < 256; ++s) acc += ks[s] * kw[tid + 255 - s];
    __syncthreads();
  }
  kout[c * L_SEQ + t] = acc + kD[c] * kc[t];
}

// -------- kv outer product, stored bf16 as [hh][col=n1*8+n2][s] --------
__global__ void kv_mul(const float* __restrict__ ksh, const float* __restrict__ vv,
                       unsigned short* __restrict__ kvbf) {
  int idx = blockIdx.x * blockDim.x + threadIdx.x;
  int s = idx & (L_SEQ - 1);
  int tmp = idx >> 11;
  int col = tmp & 63;
  int hh = tmp >> 6;
  int n1 = col >> 3, n2 = col & 7;
  float val = ksh[(hh * 8 + n1) * L_SEQ + s] * vv[(hh * 8 + n2) * L_SEQ + s];
  kvbf[idx] = f2bf(val);
}

// -------- diag-SSM: per hh, Y(2048x64) = TriToeplitz(kern_hh) x KV_hh --------
// Wave: 32 t-rows x all 64 cols = 2x4 wmma tiles. A built from LDS pair table:
// ptab[x] packs {bf16(kern[x]), bf16(kern[x-1])} so one aligned b32 LDS load
// fills one A slot; ptab[2048]=0 is the clamp target for t<s.
__global__ void __launch_bounds__(128, 1)
diag_conv(const unsigned short* __restrict__ kvbf,
          const float* __restrict__ kern,
          unsigned short* __restrict__ yconv) {
  int hh = blockIdx.x;
  int wave = threadIdx.x >> 5;
  int tbase = blockIdx.y * 128 + wave * 32;
  int lane = threadIdx.x & 31, mrow = lane & 15, hi = lane >> 4;
  __shared__ unsigned int ptab[L_SEQ + 1];
  for (int i = threadIdx.x; i < L_SEQ; i += blockDim.x) {
    unsigned int lo = f2bf(kern[hh * L_SEQ + i]);
    unsigned int h2 = (i >= 1) ? (unsigned int)f2bf(kern[hh * L_SEQ + i - 1]) : 0u;
    ptab[i] = lo | (h2 << 16);
  }
  if (threadIdx.x == 0) ptab[L_SEQ] = 0u;
  __syncthreads();
  v8f acc[2][4] = {};
  const unsigned short* bbase = kvbf + hh * 64 * L_SEQ;
  int smax = tbase + 16 + 15;                    // bound for largest M tile
  for (int s0 = 0; s0 <= smax; s0 += 32) {
    Frag b[4];
#pragma unroll
    for (int ct = 0; ct < 4; ++ct) {
      const unsigned short* brow = bbase + (ct * 16 + mrow) * L_SEQ + s0;
#pragma unroll
      for (int j = 0; j < 8; ++j)
        b[ct].u[j] = *(const unsigned int*)(brow + 2 * j + 16 * hi);
    }
#pragma unroll
    for (int mi = 0; mi < 2; ++mi) {
      int tm = tbase + 16 * mi;
      if (s0 <= tm + 15) {                       // wave-uniform: EXEC untouched
        Frag a;
#pragma unroll
        for (int j = 0; j < 8; ++j) {
          int x = tm + mrow - s0 - akk(j, hi);
          a.u[j] = ptab[(x >= 0) ? x : L_SEQ];
        }
#pragma unroll
        for (int ct = 0; ct < 4; ++ct)
          acc[mi][ct] = wmma_bf16(a.v, b[ct].v, acc[mi][ct]);
      }
    }
  }
#pragma unroll
  for (int mi = 0; mi < 2; ++mi)
#pragma unroll
    for (int ct = 0; ct < 4; ++ct) {
      unsigned int* op = (unsigned int*)(yconv + (hh * 64 + ct * 16 + mrow) * L_SEQ +
                                         tbase + 16 * mi + 8 * hi);
      uint4 pk;
      pk.x = (unsigned int)f2bf(acc[mi][ct][0]) | ((unsigned int)f2bf(acc[mi][ct][1]) << 16);
      pk.y = (unsigned int)f2bf(acc[mi][ct][2]) | ((unsigned int)f2bf(acc[mi][ct][3]) << 16);
      pk.z = (unsigned int)f2bf(acc[mi][ct][4]) | ((unsigned int)f2bf(acc[mi][ct][5]) << 16);
      pk.w = (unsigned int)f2bf(acc[mi][ct][6]) | ((unsigned int)f2bf(acc[mi][ct][7]) << 16);
      *(uint4*)op = pk;
    }
}

// -------- mul_sum: ybld[t, n2*128+hh] = sum_n1 q*(yconv + D*k'*v), bf16 --------
__global__ void combine(const float* __restrict__ q, const float* __restrict__ ksh,
                        const float* __restrict__ v,
                        const unsigned short* __restrict__ yconv,
                        const float* __restrict__ Dv,
                        unsigned short* __restrict__ ybld) {
  int idx = blockIdx.x * blockDim.x + threadIdx.x;
  int t = idx & (L_SEQ - 1);
  int tmp = idx >> 11;
  int hh = tmp & 127;
  int n2 = tmp >> 7;
  float Dh = Dv[hh];
  float vch = v[(hh * 8 + n2) * L_SEQ + t];
  float s = 0.f;
#pragma unroll
  for (int n1 = 0; n1 < 8; ++n1) {
    float yc = bf2f(yconv[(hh * 64 + n1 * 8 + n2) * L_SEQ + t]);
    float kv = ksh[(hh * 8 + n1) * L_SEQ + t] * vch;
    s += q[(hh * 8 + n1) * L_SEQ + t] * (yc + Dh * kv);
  }
  ybld[t * DMODEL + n2 * HGRP + hh] = f2bf(s);
}

// -------- out proj: out[t,e] = sum_c ybld[t,c]*Wo[e,c] + bo[e] --------
__global__ void __launch_bounds__(256, 1)
out_gemm(const unsigned short* __restrict__ ybld,
         const unsigned short* __restrict__ Wo,
         const float* __restrict__ bo, float* __restrict__ out) {
  int wave = threadIdx.x >> 5;
  int wid = blockIdx.x * 8 + wave;              // 1024 waves
  int t0 = (wid & 63) * 32;
  int e0 = (wid >> 6) * 64;
  int lane = threadIdx.x & 31, mrow = lane & 15, hi = lane >> 4;
  v8f acc[2][4] = {};
  const unsigned short* ar[2];
  const unsigned short* br[4];
#pragma unroll
  for (int i = 0; i < 2; ++i) ar[i] = ybld + (t0 + 16 * i + mrow) * DMODEL;
#pragma unroll
  for (int i = 0; i < 4; ++i) br[i] = Wo + (e0 + 16 * i + mrow) * DMODEL;
  for (int k0 = 0; k0 < DMODEL; k0 += 32) {
    __builtin_prefetch(ar[0] + k0 + 256);
    __builtin_prefetch(br[0] + k0 + 256);
    Frag a[2], b[4];
#pragma unroll
    for (int i = 0; i < 2; ++i)
#pragma unroll
      for (int j = 0; j < 8; ++j)
        a[i].u[j] = *(const unsigned int*)(ar[i] + k0 + akk(j, hi));
#pragma unroll
    for (int i = 0; i < 4; ++i)
#pragma unroll
      for (int j = 0; j < 8; ++j)
        b[i].u[j] = *(const unsigned int*)(br[i] + k0 + 2 * j + 16 * hi);
#pragma unroll
    for (int mi = 0; mi < 2; ++mi)
#pragma unroll
      for (int ct = 0; ct < 4; ++ct)
        acc[mi][ct] = wmma_bf16(a[mi].v, b[ct].v, acc[mi][ct]);
  }
#pragma unroll
  for (int ct = 0; ct < 4; ++ct) {
    float bb = bo[e0 + 16 * ct + mrow];
#pragma unroll
    for (int mi = 0; mi < 2; ++mi)
#pragma unroll
      for (int r = 0; r < 8; ++r)
        out[(t0 + 16 * mi + 8 * hi + r) * DMODEL + e0 + 16 * ct + mrow] =
            acc[mi][ct][r] + bb;
  }
}

extern "C" void kernel_launch(void* const* d_in, const int* in_sizes, int n_in,
                              void* d_out, int out_size, void* d_ws, size_t ws_size,
                              hipStream_t stream) {
  const float* u = (const float*)d_in[0];
  const float* Wq = (const float*)d_in[1];
  const float* bq = (const float*)d_in[2];
  const float* Wk = (const float*)d_in[3];
  const float* bk = (const float*)d_in[4];
  const float* Wv = (const float*)d_in[5];
  const float* bv = (const float*)d_in[6];
  const float* kernk = (const float*)d_in[7];
  const float* kD = (const float*)d_in[8];
  const float* kern = (const float*)d_in[9];
  const float* Dv = (const float*)d_in[10];
  const float* Wo = (const float*)d_in[11];
  const float* bo = (const float*)d_in[12];

  char* ws = (char*)d_ws;
  size_t off = 0;
  auto carve = [&](size_t bytes) -> char* {
    char* p = ws + off;
    off += (bytes + 255) & ~(size_t)255;
    return p;
  };
  const size_t LD = (size_t)L_SEQ * DMODEL;
  unsigned short* u_bf = (unsigned short*)carve(LD * 2);
  unsigned short* Wq_bf = (unsigned short*)carve((size_t)DMODEL * DMODEL * 2);
  unsigned short* Wk_bf = (unsigned short*)carve((size_t)DMODEL * DMODEL * 2);
  unsigned short* Wv_bf = (unsigned short*)carve((size_t)DMODEL * DMODEL * 2);
  unsigned short* Wo_bf = (unsigned short*)carve((size_t)DMODEL * DMODEL * 2);
  float* qb = (float*)carve(LD * 4);
  float* kb = (float*)carve(LD * 4);
  float* vb = (float*)carve(LD * 4);
  float* kshift = (float*)carve(LD * 4);
  unsigned short* kv_bf = (unsigned short*)carve((size_t)HGRP * 64 * L_SEQ * 2);
  unsigned short* yconv = (unsigned short*)carve((size_t)HGRP * 64 * L_SEQ * 2);
  unsigned short* ybld = (unsigned short*)carve(LD * 2);
  // total ~112 MB

  cvt_bf16<<<512, 256, 0, stream>>>(u, u_bf, L_SEQ * DMODEL);
  cvt_bf16<<<512, 256, 0, stream>>>(Wq, Wq_bf, DMODEL * DMODEL);
  cvt_bf16<<<512, 256, 0, stream>>>(Wk, Wk_bf, DMODEL * DMODEL);
  cvt_bf16<<<512, 256, 0, stream>>>(Wv, Wv_bf, DMODEL * DMODEL);
  cvt_bf16<<<512, 256, 0, stream>>>(Wo, Wo_bf, DMODEL * DMODEL);

  qkv_gemm<<<dim3(128, 3), 256, 0, stream>>>(u_bf, Wq_bf, Wk_bf, Wv_bf, bq, bk, bv,
                                             qb, kb, vb);
  shift_conv<<<dim3(1024, 8), 256, 0, stream>>>(kb, kernk, kD, kshift);
  kv_mul<<<(HGRP * 64 * L_SEQ) / 256, 256, 0, stream>>>(kshift, vb, kv_bf);
  diag_conv<<<dim3(128, 16), 128, 0, stream>>>(kv_bf, kern, yconv);
  combine<<<(L_SEQ * HGRP * NHEAD) / 256, 256, 0, stream>>>(qb, kshift, vb, yconv,
                                                            Dv, ybld);
  out_gemm<<<128, 256, 0, stream>>>(ybld, Wo_bf, bo, (float*)d_out);
}
